// NodeModel_35012573397768
// MI455X (gfx1250) — compile-verified, matrix operand order checked by reading, not run on hardware
//
#include <hip/hip_runtime.h>
#include <hip/hip_bf16.h>

// Problem constants (from reference)
#define DFEAT 128
#define OUTF  128
#define IN1   512   // 4*D  : edge MLP input
#define IN2   384   // D+OUT+D : node MLP input

typedef __attribute__((ext_vector_type(16))) __bf16 v16bf;
typedef __attribute__((ext_vector_type(8)))  float  v8f;
typedef __attribute__((ext_vector_type(4)))  float  f32x4;          // clang ext vector:
typedef __attribute__((ext_vector_type(4)))  unsigned short u16x4;  // OK for NT builtins

union BF16x16 { uint4 q[2]; v16bf v; };

__device__ __forceinline__ unsigned short f32_to_bf16(float f) {
  union { float f; unsigned int u; } c; c.f = f;
  unsigned int u = c.u;
  u += 0x7FFFu + ((u >> 16) & 1u);     // round-to-nearest-even
  return (unsigned short)(u >> 16);
}

__device__ __forceinline__ u16x4 cvt4(f32x4 f) {
  u16x4 h;
  h.x = f32_to_bf16(f.x); h.y = f32_to_bf16(f.y);
  h.z = f32_to_bf16(f.z); h.w = f32_to_bf16(f.w);
  return h;
}

// ---------------------------------------------------------------------------
// prep: zero aggregation buffer, convert W1/W2 fp32 -> bf16
// ---------------------------------------------------------------------------
__global__ void prep_kernel(const float* __restrict__ W1,
                            const float* __restrict__ W2,
                            unsigned short* __restrict__ w1bf,
                            unsigned short* __restrict__ w2bf,
                            float* __restrict__ agg, int nAgg) {
  int i = blockIdx.x * 256 + threadIdx.x;
  if (i < nAgg) agg[i] = 0.0f;
  if (i < OUTF * IN1) w1bf[i] = f32_to_bf16(W1[i]);
  if (i < OUTF * IN2) w2bf[i] = f32_to_bf16(W2[i]);
}

// ---------------------------------------------------------------------------
// edge kernel: 64 edges per block, 8 waves; wave w computes out cols [16w,16w+16)
// fused edge-MLP (bf16 WMMA, f32 accum) + ReLU + atomic scatter-add by dst
// ---------------------------------------------------------------------------
__global__ __launch_bounds__(256) void edge_kernel(
    const float* __restrict__ x, const int* __restrict__ edge_index,
    const float* __restrict__ edge_attr, const float* __restrict__ u,
    const int* __restrict__ batch, const unsigned short* __restrict__ w1bf,
    const float* __restrict__ b1, float* __restrict__ agg, int E) {
  __shared__ alignas(16) unsigned short sA[64][IN1 + 8];  // +8 bf16 pad -> bank spread
  __shared__ int s_dst[64];

  const int e0 = blockIdx.x * 64;
  const int t  = threadIdx.x;

  // Stage 64 x 512 edge-input tile to LDS in bf16. 64*128 = 8192 float4 slots.
  for (int i = 0; i < 32; ++i) {
    int flat   = t + i * 256;          // 0..8191
    int e      = flat >> 7;            // 128 float4 per edge row
    int pos    = flat & 127;
    int seg    = pos >> 5;             // 0:x[dst] 1:x[src] 2:edge_attr 3:u[batch[dst]]
    int within = (pos & 31) << 2;      // float offset 0..124
    int eg     = e0 + e; if (eg >= E) eg = E - 1;
    f32x4 f;
    if (seg == 2) {
      // streamed exactly once: non-temporal so it doesn't evict x/agg from L2
      f = __builtin_nontemporal_load(
              (const f32x4*)(edge_attr + (size_t)eg * DFEAT + within));
    } else {
      const float* srcp;
      if (seg == 0)      srcp = x + (size_t)edge_index[E + eg] * DFEAT + within;
      else if (seg == 1) srcp = x + (size_t)edge_index[eg] * DFEAT + within;
      else               srcp = u + (size_t)batch[edge_index[E + eg]] * DFEAT + within;
      f = *(const f32x4*)srcp;
    }
    *(u16x4*)&sA[e][pos << 2] = cvt4(f);
  }
  if (t < 64) { int eg = e0 + t; s_dst[t] = edge_index[E + (eg < E ? eg : E - 1)]; }
  __syncthreads();

  const int wave = t >> 5;
  const int lane = t & 31;
  const int rc   = lane & 15;          // A row within 16-tile == B column within 16-tile
  const int kh   = lane >> 4;          // K-half select (ISA 16-bit fragment layout)
  const int colg = (wave << 4) + rc;   // global output column 0..127

  v8f acc[4] = {};
  const unsigned short* wrow = w1bf + (size_t)colg * IN1;  // B col = W1 row (x @ W1^T)

  for (int kt = 0; kt < IN1 / 32; ++kt) {
    int kb = kt * 32 + kh * 8;
    BF16x16 b;
    b.q[0] = *(const uint4*)&wrow[kb];
    b.q[1] = *(const uint4*)&wrow[kb + 16];
#pragma unroll
    for (int mt = 0; mt < 4; ++mt) {   // reuse B fragment across 4 M-subtiles
      BF16x16 a;
      a.q[0] = *(const uint4*)&sA[(mt << 4) + rc][kb];
      a.q[1] = *(const uint4*)&sA[(mt << 4) + rc][kb + 16];
      acc[mt] = __builtin_amdgcn_wmma_f32_16x16x32_bf16(
          false, a.v, false, b.v, (short)0, acc[mt], false, false);
    }
  }

  const float bias = b1[colg];
  if (e0 + 64 <= E) {
    // full tile: straight-line atomic scatter, no per-element exec masking
#pragma unroll
    for (int mt = 0; mt < 4; ++mt) {
#pragma unroll
      for (int r = 0; r < 8; ++r) {
        float v = fmaxf(acc[mt][r] + bias, 0.0f);
        int erow = (mt << 4) + (kh << 3) + r;   // C layout: lanes16-31 -> M = 8+r
        atomicAdd(&agg[(size_t)s_dst[erow] * OUTF + colg], v);
      }
    }
  } else {
#pragma unroll
    for (int mt = 0; mt < 4; ++mt) {
#pragma unroll
      for (int r = 0; r < 8; ++r) {
        float v = fmaxf(acc[mt][r] + bias, 0.0f);
        int erow = (mt << 4) + (kh << 3) + r;
        if (e0 + erow < E)
          atomicAdd(&agg[(size_t)s_dst[erow] * OUTF + colg], v);
      }
    }
  }
}

// ---------------------------------------------------------------------------
// node kernel: 64 nodes per block, fused node-MLP (bf16 WMMA) + ReLU, NT store
// ---------------------------------------------------------------------------
__global__ __launch_bounds__(256) void node_kernel(
    const float* __restrict__ x, const float* __restrict__ agg,
    const float* __restrict__ u, const int* __restrict__ batch,
    const unsigned short* __restrict__ w2bf, const float* __restrict__ b2,
    float* __restrict__ out, int N) {
  __shared__ alignas(16) unsigned short sA[64][IN2 + 8];

  const int n0 = blockIdx.x * 64;
  const int t  = threadIdx.x;

  // Stage 64 x 384 node-input tile: 64*96 = 6144 float4 slots.
  for (int i = 0; i < 24; ++i) {
    int flat   = t + i * 256;          // 0..6143
    int e      = flat / 96;
    int pos    = flat % 96;
    int seg    = pos >> 5;             // 0:x 1:agg 2:u[batch]
    int within = (pos & 31) << 2;
    int ng     = n0 + e; if (ng >= N) ng = N - 1;
    const float* srcp;
    if (seg == 0)      srcp = x + (size_t)ng * DFEAT + within;
    else if (seg == 1) srcp = agg + (size_t)ng * OUTF + within;
    else               srcp = u + (size_t)batch[ng] * DFEAT + within;
    *(u16x4*)&sA[e][pos << 2] = cvt4(*(const f32x4*)srcp);
  }
  __syncthreads();

  const int wave = t >> 5;
  const int lane = t & 31;
  const int rc   = lane & 15;
  const int kh   = lane >> 4;
  const int colg = (wave << 4) + rc;

  v8f acc[4] = {};
  const unsigned short* wrow = w2bf + (size_t)colg * IN2;

  for (int kt = 0; kt < IN2 / 32; ++kt) {
    int kb = kt * 32 + kh * 8;
    BF16x16 b;
    b.q[0] = *(const uint4*)&wrow[kb];
    b.q[1] = *(const uint4*)&wrow[kb + 16];
#pragma unroll
    for (int mt = 0; mt < 4; ++mt) {
      BF16x16 a;
      a.q[0] = *(const uint4*)&sA[(mt << 4) + rc][kb];
      a.q[1] = *(const uint4*)&sA[(mt << 4) + rc][kb + 16];
      acc[mt] = __builtin_amdgcn_wmma_f32_16x16x32_bf16(
          false, a.v, false, b.v, (short)0, acc[mt], false, false);
    }
  }

  const float bias = b2[colg];
  if (n0 + 64 <= N) {
#pragma unroll
    for (int mt = 0; mt < 4; ++mt) {
#pragma unroll
      for (int r = 0; r < 8; ++r) {
        float v = fmaxf(acc[mt][r] + bias, 0.0f);
        int nrow = (mt << 4) + (kh << 3) + r;
        // write-once output: non-temporal store
        __builtin_nontemporal_store(v, &out[(size_t)(n0 + nrow) * OUTF + colg]);
      }
    }
  } else {
#pragma unroll
    for (int mt = 0; mt < 4; ++mt) {
#pragma unroll
      for (int r = 0; r < 8; ++r) {
        float v = fmaxf(acc[mt][r] + bias, 0.0f);
        int nrow = (mt << 4) + (kh << 3) + r;
        if (n0 + nrow < N)
          __builtin_nontemporal_store(v, &out[(size_t)(n0 + nrow) * OUTF + colg]);
      }
    }
  }
}

// ---------------------------------------------------------------------------
extern "C" void kernel_launch(void* const* d_in, const int* in_sizes, int n_in,
                              void* d_out, int out_size, void* d_ws, size_t ws_size,
                              hipStream_t stream) {
  const float* x          = (const float*)d_in[0];
  const int*   edge_index = (const int*)d_in[1];   // [2, E]
  const float* edge_attr  = (const float*)d_in[2];
  const float* u          = (const float*)d_in[3];
  const int*   batch      = (const int*)d_in[4];
  const float* W1         = (const float*)d_in[5];
  const float* b1         = (const float*)d_in[6];
  const float* W2         = (const float*)d_in[7];
  const float* b2         = (const float*)d_in[8];
  float*       out        = (float*)d_out;

  const int N = in_sizes[4];          // batch has one entry per node
  const int E = in_sizes[1] / 2;      // edge_index is [2, E]

  // Workspace layout: [agg f32 N*OUTF][W1 bf16][W2 bf16]
  float* agg = (float*)d_ws;
  unsigned short* w1bf = (unsigned short*)((char*)d_ws + (size_t)N * OUTF * sizeof(float));
  unsigned short* w2bf = w1bf + OUTF * IN1;

  const int nAgg = N * OUTF;
  prep_kernel<<<(nAgg + 255) / 256, 256, 0, stream>>>(W1, W2, w1bf, w2bf, agg, nAgg);
  edge_kernel<<<(E + 63) / 64, 256, 0, stream>>>(x, edge_index, edge_attr, u, batch,
                                                 w1bf, b1, agg, E);
  node_kernel<<<(N + 63) / 64, 256, 0, stream>>>(x, agg, u, batch, w2bf, b2, out, N);
}
